// FusedExpertsNetwork_1726576856628
// MI455X (gfx1250) — matrix operand order repaired and checked by reading, not compiled, not run on hardware
//
#include <hip/hip_runtime.h>

// Problem dims (fixed by the reference)
#define E_ 8
#define C_ 2048
#define M_ 1024
#define H_ 4096

// Tile config: 128x128x32 block tile, 256 threads (8 wave32, 4x2 wave grid),
// each wave computes a 32x64 sub-tile = 2x4 WMMA 16x16 accumulators.
#define BM 128
#define BN 128
#define BK 32
#define PITCH 40   // LDS row pitch in bf16 elements (80B: 16B-aligned, bank-spread)

typedef __attribute__((ext_vector_type(16))) __bf16 v16bf;
typedef __attribute__((ext_vector_type(8)))  float  v8f;

union Frag { uint4 u[2]; v16bf v; };

#if __has_builtin(__builtin_amdgcn_cvt_pk_bf16_f32)
__device__ __forceinline__ unsigned pack_bf16(float a, float b) {
  auto r = __builtin_amdgcn_cvt_pk_bf16_f32(a, b);
  union { decltype(r) v; unsigned u; } c;
  c.v = r;
  return c.u;
}
#else
__device__ __forceinline__ unsigned short f2bf_1(float f) {
  union { float f; unsigned u; } c; c.f = f;
  unsigned r = c.u + 0x7FFFu + ((c.u >> 16) & 1u);   // RNE
  return (unsigned short)(r >> 16);
}
__device__ __forceinline__ unsigned pack_bf16(float a, float b) {
  return (unsigned)f2bf_1(a) | ((unsigned)f2bf_1(b) << 16);
}
#endif

__device__ __forceinline__ unsigned short f2bf(float f) {
  return (unsigned short)(pack_bf16(f, f) & 0xFFFFu);
}

// ---------- staging helpers ----------

// 128x32 fp32 tile (row stride in floats) -> LDS bf16 [128][PITCH]
__device__ __forceinline__ void stage_cvt(const float* __restrict__ src, int srcStride,
                                          unsigned short* dst, int tid) {
  #pragma unroll
  for (int i = 0; i < 4; ++i) {
    int idx = tid + i * 256;
    int r = idx >> 3, c4 = (idx & 7) << 2;
    const float4 v = *(const float4*)(src + (size_t)r * srcStride + c4);
    uint2 p;
    p.x = pack_bf16(v.x, v.y);
    p.y = pack_bf16(v.z, v.w);
    *(uint2*)(&dst[r * PITCH + c4]) = p;
  }
}

// 128x32 bf16 tile (row stride in elements) -> LDS, straight b128 copy
__device__ __forceinline__ void stage_copy(const unsigned short* __restrict__ src, int srcStride,
                                           unsigned short* dst, int tid) {
  #pragma unroll
  for (int i = 0; i < 2; ++i) {
    int idx = tid + i * 256;
    int r = idx >> 2, c8 = (idx & 3) << 3;
    *(uint4*)(&dst[r * PITCH + c8]) = *(const uint4*)(src + (size_t)r * srcStride + c8);
  }
}

// 32x128 fp32 tile (global [k][n]) transposed into LDS [n][k] bf16; (k,k+1)
// pairs packed so each store is an aligned ds_store_b32.
__device__ __forceinline__ void stage_cvt_tr(const float* __restrict__ src, int srcStride,
                                             unsigned short* dst, int tid) {
  #pragma unroll
  for (int i = 0; i < 2; ++i) {
    int idx = tid + i * 256;
    int k  = (idx >> 5) << 1;
    int c4 = (idx & 31) << 2;
    const float4 v0 = *(const float4*)(src + (size_t)k * srcStride + c4);
    const float4 v1 = *(const float4*)(src + (size_t)(k + 1) * srcStride + c4);
    *(unsigned*)(&dst[(c4 + 0) * PITCH + k]) = pack_bf16(v0.x, v1.x);
    *(unsigned*)(&dst[(c4 + 1) * PITCH + k]) = pack_bf16(v0.y, v1.y);
    *(unsigned*)(&dst[(c4 + 2) * PITCH + k]) = pack_bf16(v0.z, v1.z);
    *(unsigned*)(&dst[(c4 + 3) * PITCH + k]) = pack_bf16(v0.w, v1.w);
  }
}

// One BK=32 step: fragment loads per ISA layouts, 8 WMMAs per wave.
__device__ __forceinline__ void mma_step(const unsigned short* As, const unsigned short* Bs,
                                         int wm, int wn, int hl, int l16, v8f acc[2][4]) {
  Frag a[2], b[4];
  #pragma unroll
  for (int mi = 0; mi < 2; ++mi) {
    const unsigned short* base = &As[(wm * 32 + mi * 16 + l16) * PITCH];
    a[mi].u[0] = *(const uint4*)(base + hl * 8);
    a[mi].u[1] = *(const uint4*)(base + 16 + hl * 8);
  }
  #pragma unroll
  for (int ni = 0; ni < 4; ++ni) {
    const unsigned short* base = &Bs[(wn * 64 + ni * 16 + l16) * PITCH];
    b[ni].u[0] = *(const uint4*)(base + hl * 8);
    b[ni].u[1] = *(const uint4*)(base + 16 + hl * 8);
  }
  #pragma unroll
  for (int mi = 0; mi < 2; ++mi)
    #pragma unroll
    for (int ni = 0; ni < 4; ++ni)
      acc[mi][ni] = __builtin_amdgcn_wmma_f32_16x16x32_bf16(
          false, a[mi].v, false, b[ni].v, (short)0, acc[mi][ni], false, false);
}

// ---------- one-time conversion kernels (fast path) ----------

// Elementwise fp32 -> bf16, 8 elements/thread.
__global__ __launch_bounds__(256) void cvt_f32_bf16(const float* __restrict__ src,
                                                    unsigned short* __restrict__ dst) {
  long i = ((long)blockIdx.x * 256 + threadIdx.x) * 8;
  const float4 v0 = *(const float4*)(src + i);
  const float4 v1 = *(const float4*)(src + i + 4);
  uint4 p;
  p.x = pack_bf16(v0.x, v0.y);
  p.y = pack_bf16(v0.z, v0.w);
  p.z = pack_bf16(v1.x, v1.y);
  p.w = pack_bf16(v1.z, v1.w);
  *(uint4*)(dst + i) = p;
}

// w2 [E,H,M] fp32 -> w2t [E,M,H] bf16, 64x64 LDS tile transpose.
#define TT 64
#define TP 72   // pad: 72*2 = 144 bytes/row (multiple of 16)
__global__ __launch_bounds__(256) void cvt_tr_w2(const float* __restrict__ w2,
                                                 unsigned short* __restrict__ w2t) {
  __shared__ __align__(16) unsigned short t[TT * TP];
  const int e  = blockIdx.z;
  const int h0 = blockIdx.x * TT;
  const int m0 = blockIdx.y * TT;
  const float* src = w2 + (size_t)e * H_ * M_ + (size_t)h0 * M_ + m0;
  unsigned short* dst = w2t + (size_t)e * M_ * H_ + (size_t)m0 * H_ + h0;
  const int tid = threadIdx.x;
  #pragma unroll
  for (int i = 0; i < 4; ++i) {
    int idx = tid + i * 256;            // 1024 float4: 64 h-rows x 16 m-quads
    int h = idx >> 4, m4 = (idx & 15) << 2;
    const float4 v = *(const float4*)(src + (size_t)h * M_ + m4);
    t[(m4 + 0) * TP + h] = f2bf(v.x);
    t[(m4 + 1) * TP + h] = f2bf(v.y);
    t[(m4 + 2) * TP + h] = f2bf(v.z);
    t[(m4 + 3) * TP + h] = f2bf(v.w);
  }
  __syncthreads();
  #pragma unroll
  for (int i = 0; i < 2; ++i) {
    int idx = tid + i * 256;            // 512 uint4: 64 m-rows x 8 h-octets
    int m = idx >> 3, h8 = (idx & 7) << 3;
    *(uint4*)(dst + (size_t)m * H_ + h8) = *(const uint4*)(&t[m * TP + h8]);
  }
}

// ---------------- GEMM1: act = relu(A @ W1^T + b1), bf16 output ----------------
template <bool PRE>
__global__ __launch_bounds__(256) void ffn_gemm1(
    const void* __restrict__ xin,   // PRE ? bf16 [E,C,M] : fp32 [E,C,M]
    const void* __restrict__ w1in,  // PRE ? bf16 [E,H,M] : fp32 [E,H,M]
    const float* __restrict__ b1,
    unsigned short* __restrict__ act)
{
  __shared__ __align__(16) unsigned short As0[BM * PITCH], As1[BM * PITCH];
  __shared__ __align__(16) unsigned short Bs0[BN * PITCH], Bs1[BN * PITCH];

  const int e  = blockIdx.z;
  const int m0 = blockIdx.y * BM;
  const int n0 = blockIdx.x * BN;
  const float*          aF = (const float*)xin          + (size_t)e * C_ * M_ + (size_t)m0 * M_;
  const float*          bF = (const float*)w1in         + (size_t)e * H_ * M_ + (size_t)n0 * M_;
  const unsigned short* aH = (const unsigned short*)xin + (size_t)e * C_ * M_ + (size_t)m0 * M_;
  const unsigned short* bH = (const unsigned short*)w1in+ (size_t)e * H_ * M_ + (size_t)n0 * M_;
  const float* be = b1 + (size_t)e * H_;
  unsigned short* ae = act + (size_t)e * C_ * H_;

  const int tid  = threadIdx.x;
  const int lane = tid & 31;
  const int wid  = tid >> 5;
  const int wm = wid & 3, wn = wid >> 2;
  const int hl = lane >> 4, l16 = lane & 15;

  v8f acc[2][4] = {};

  if constexpr (PRE) { stage_copy(aH, M_, As0, tid); stage_copy(bH, M_, Bs0, tid); }
  else               { stage_cvt (aF, M_, As0, tid); stage_cvt (bF, M_, Bs0, tid); }

  for (int k0 = 0; k0 < M_; k0 += 2 * BK) {
    __syncthreads();
    if constexpr (PRE) { stage_copy(aH + k0 + BK, M_, As1, tid); stage_copy(bH + k0 + BK, M_, Bs1, tid); }
    else               { stage_cvt (aF + k0 + BK, M_, As1, tid); stage_cvt (bF + k0 + BK, M_, Bs1, tid); }
    mma_step(As0, Bs0, wm, wn, hl, l16, acc);
    __syncthreads();
    if (k0 + 2 * BK < M_) {
      if constexpr (PRE) { stage_copy(aH + k0 + 2*BK, M_, As0, tid); stage_copy(bH + k0 + 2*BK, M_, Bs0, tid); }
      else               { stage_cvt (aF + k0 + 2*BK, M_, As0, tid); stage_cvt (bF + k0 + 2*BK, M_, Bs0, tid); }
    }
    mma_step(As1, Bs1, wm, wn, hl, l16, acc);
  }

  #pragma unroll
  for (int ni = 0; ni < 4; ++ni) {
    const int h = n0 + wn * 64 + ni * 16 + l16;
    const float bias = be[h];
    #pragma unroll
    for (int mi = 0; mi < 2; ++mi) {
      const int mrow = m0 + wm * 32 + mi * 16 + hl * 8;
      #pragma unroll
      for (int i = 0; i < 8; ++i) {
        float v = acc[mi][ni][i] + bias;
        v = v > 0.f ? v : 0.f;
        ae[(size_t)(mrow + i) * H_ + h] = f2bf(v);
      }
    }
  }
}

// ---------------- GEMM2: out = act @ W2 + b2, fp32 output ----------------
template <bool PRE>
__global__ __launch_bounds__(256) void ffn_gemm2(
    const unsigned short* __restrict__ act, // [E,C,H] bf16
    const void* __restrict__ w2in,          // PRE ? bf16 [E,M,H] (pre-transposed) : fp32 [E,H,M]
    const float* __restrict__ b2,
    float* __restrict__ out)
{
  __shared__ __align__(16) unsigned short As0[BM * PITCH], As1[BM * PITCH];
  __shared__ __align__(16) unsigned short Bs0[BN * PITCH], Bs1[BN * PITCH];

  const int e  = blockIdx.z;
  const int m0 = blockIdx.y * BM;
  const int n0 = blockIdx.x * BN;
  const unsigned short* aSrc = act + (size_t)e * C_ * H_ + (size_t)m0 * H_;
  const float*          bF = (const float*)w2in          + (size_t)e * H_ * M_ + n0;
  const unsigned short* bH = (const unsigned short*)w2in + (size_t)e * M_ * H_ + (size_t)n0 * H_;
  const float* be = b2 + (size_t)e * M_;
  float* oe = out + (size_t)e * C_ * M_;

  const int tid  = threadIdx.x;
  const int lane = tid & 31;
  const int wid  = tid >> 5;
  const int wm = wid & 3, wn = wid >> 2;
  const int hl = lane >> 4, l16 = lane & 15;

  v8f acc[2][4] = {};

  stage_copy(aSrc, H_, As0, tid);
  if constexpr (PRE) stage_copy(bH, H_, Bs0, tid);
  else               stage_cvt_tr(bF, M_, Bs0, tid);

  for (int k0 = 0; k0 < H_; k0 += 2 * BK) {
    __syncthreads();
    stage_copy(aSrc + k0 + BK, H_, As1, tid);
    if constexpr (PRE) stage_copy(bH + k0 + BK, H_, Bs1, tid);
    else               stage_cvt_tr(bF + (size_t)(k0 + BK) * M_, M_, Bs1, tid);
    mma_step(As0, Bs0, wm, wn, hl, l16, acc);
    __syncthreads();
    if (k0 + 2 * BK < H_) {
      stage_copy(aSrc + k0 + 2 * BK, H_, As0, tid);
      if constexpr (PRE) stage_copy(bH + k0 + 2 * BK, H_, Bs0, tid);
      else               stage_cvt_tr(bF + (size_t)(k0 + 2 * BK) * M_, M_, Bs0, tid);
    }
    mma_step(As1, Bs1, wm, wn, hl, l16, acc);
  }

  #pragma unroll
  for (int ni = 0; ni < 4; ++ni) {
    const int o = n0 + wn * 64 + ni * 16 + l16;
    const float bias = be[o];
    #pragma unroll
    for (int mi = 0; mi < 2; ++mi) {
      const int mrow = m0 + wm * 32 + mi * 16 + hl * 8;
      #pragma unroll
      for (int i = 0; i < 8; ++i)
        oe[(size_t)(mrow + i) * M_ + o] = acc[mi][ni][i] + bias;
    }
  }
}

extern "C" void kernel_launch(void* const* d_in, const int* in_sizes, int n_in,
                              void* d_out, int out_size, void* d_ws, size_t ws_size,
                              hipStream_t stream) {
  (void)in_sizes; (void)n_in; (void)out_size;
  const float* x  = (const float*)d_in[0];
  const float* w1 = (const float*)d_in[1];
  const float* b1 = (const float*)d_in[2];
  const float* w2 = (const float*)d_in[3];
  const float* b2 = (const float*)d_in[4];
  float* out = (float*)d_out;

  const size_t actE = (size_t)E_ * C_ * H_;   // 67.1M bf16 = 128 MB
  const size_t xE   = (size_t)E_ * C_ * M_;   //  32 MB
  const size_t wE   = (size_t)E_ * H_ * M_;   //  64 MB each
  unsigned short* act = (unsigned short*)d_ws;

  dim3 blk(256);
  dim3 g1(H_ / BN, C_ / BM, E_);   // 32 x 16 x 8
  dim3 g2(M_ / BN, C_ / BM, E_);   // 8 x 16 x 8

  if (ws_size >= (actE + xE + 2 * wE) * sizeof(unsigned short)) {
    // Fast path: one-time bf16 conversion (+ W2 transpose), copy-only GEMM staging.
    unsigned short* xbf  = act  + actE;
    unsigned short* w1bf = xbf  + xE;
    unsigned short* w2t  = w1bf + wE;
    cvt_f32_bf16<<<dim3((unsigned)(xE / (8 * 256))), blk, 0, stream>>>(x,  xbf);
    cvt_f32_bf16<<<dim3((unsigned)(wE / (8 * 256))), blk, 0, stream>>>(w1, w1bf);
    cvt_tr_w2  <<<dim3(H_ / TT, M_ / TT, E_), blk, 0, stream>>>(w2, w2t);
    ffn_gemm1<true><<<g1, blk, 0, stream>>>(xbf, w1bf, b1, act);
    ffn_gemm2<true><<<g2, blk, 0, stream>>>(act, w2t, b2, out);
  } else {
    // Fallback: convert during staging (needs only the 128 MB activation).
    ffn_gemm1<false><<<g1, blk, 0, stream>>>(x, w1, b1, act);
    ffn_gemm2<false><<<g2, blk, 0, stream>>>(act, w2, b2, out);
  }
}
